// OS_CNNGAT_27582279975626
// MI455X (gfx1250) — compile-verified, matrix operand order checked by reading, not run on hardware
//
#include <hip/hip_runtime.h>

#define Nn 4096
#define Ll 128
#define Dd 256
#define NCLASSc 10
#define EPSf 1e-5f

typedef float v8f   __attribute__((ext_vector_type(8)));
typedef float f32x4 __attribute__((ext_vector_type(4)));
typedef float f32x2 __attribute__((ext_vector_type(2)));
typedef unsigned int u32x4 __attribute__((ext_vector_type(4)));
typedef int i32x4 __attribute__((ext_vector_type(4)));
typedef int i32x8 __attribute__((ext_vector_type(8)));
typedef __bf16 v16bf __attribute__((ext_vector_type(16)));

static __device__ __forceinline__ unsigned short f2bf(float f) {
  unsigned int u = __float_as_uint(f);
  unsigned int r = (u + 0x7FFFu + ((u >> 16) & 1u)) >> 16; // RNE
  return (unsigned short)r;
}

static __device__ __forceinline__ float leaky(float z) {
  return (z >= 0.f) ? z : 0.01f * z;
}

// ---------------- Layer 0: tiny conv (cin=1), bias dropped (cancels in BN) ----
__global__ void l0_conv(const float* __restrict__ x,
                        const float* __restrict__ w0, const float* __restrict__ w1,
                        const float* __restrict__ w2, const float* __restrict__ w3,
                        float* __restrict__ y) {
  __shared__ float xs[134];
  __shared__ float wsm[1024];
  const int n = blockIdx.x, tid = threadIdx.x; // 128 threads
  xs[tid + 3] = x[(size_t)n * Ll + tid];
  if (tid < 3) { xs[tid] = 0.f; xs[131 + tid] = 0.f; }
  const float* wp[4] = {w0, w1, w2, w3};
  const int off[4] = {0, 64, 256, 576};
  const int ks[4] = {1, 3, 5, 7};
  for (int ci = 0; ci < 4; ci++)
    for (int idx = tid; idx < 64 * ks[ci]; idx += 128) wsm[off[ci] + idx] = wp[ci][idx];
  __syncthreads();
  const int l = tid;
  for (int ci = 0; ci < 4; ci++) {
    int k = ks[ci], p = (k - 1) >> 1, o = off[ci];
    for (int co = 0; co < 64; co++) {
      float acc = 0.f;
      for (int t = 0; t < k; t++) acc = fmaf(wsm[o + co * k + t], xs[l + t - p + 3], acc);
      y[(((size_t)n * Dd + ci * 64 + co) << 7) + l] = acc;
    }
  }
}

// ---------------- per-channel sum / sumsq over (N,L) ------------------------
__global__ void bn_stats(const float* __restrict__ y, float* __restrict__ stats) {
  __shared__ float rs[256], rq[256];
  const int c = blockIdx.x, tid = threadIdx.x;
  float s = 0.f, q = 0.f;
  for (int nidx = tid; nidx < Nn; nidx += 256) {
    const f32x4* p = (const f32x4*)(y + (((size_t)nidx * Dd + c) << 7));
    for (int l4 = 0; l4 < 32; l4++) {
      f32x4 v = p[l4];
      s += v.x + v.y + v.z + v.w;
      q += v.x * v.x + v.y * v.y + v.z * v.z + v.w * v.w;
    }
  }
  rs[tid] = s; rq[tid] = q; __syncthreads();
  for (int off = 128; off > 0; off >>= 1) {
    if (tid < off) { rs[tid] += rs[tid + off]; rq[tid] += rq[tid + off]; }
    __syncthreads();
  }
  if (tid == 0) { stats[c] = rs[0]; stats[Dd + c] = rq[0]; }
}

// ---------------- BN+ReLU, f32 [n][c][l] -> bf16 [n][l][c] ------------------
__global__ void l0_apply(const float* __restrict__ y, const float* __restrict__ stats,
                         const float* g0, const float* g1, const float* g2, const float* g3,
                         const float* b0, const float* b1, const float* b2, const float* b3,
                         unsigned short* __restrict__ h0) {
  __shared__ float st[32 * 129];
  __shared__ float sscale[32], sshift[32];
  const int n = blockIdx.x, tid = threadIdx.x;
  const float cnt = (float)(Nn * Ll);
  for (int cg = 0; cg < 8; ++cg) {
    __syncthreads();
    if (tid < 32) {
      int c = cg * 32 + tid;
      float mean = stats[c] / cnt;
      float var = stats[Dd + c] / cnt - mean * mean;
      int ci = c >> 6, cidx = c & 63;
      const float* gp = (ci == 0) ? g0 : (ci == 1) ? g1 : (ci == 2) ? g2 : g3;
      const float* bp = (ci == 0) ? b0 : (ci == 1) ? b1 : (ci == 2) ? b2 : b3;
      float sc = gp[cidx] * rsqrtf(var + EPSf);
      sscale[tid] = sc;
      sshift[tid] = bp[cidx] - mean * sc;
    }
    for (int idx = tid; idx < 32 * 128; idx += 256) {
      int cl = idx >> 7, l = idx & 127;
      st[cl * 129 + l] = y[(((size_t)n * Dd + cg * 32 + cl) << 7) + l];
    }
    __syncthreads();
    for (int idx = tid; idx < 32 * 128; idx += 256) {
      int l = idx >> 5, cl = idx & 31;
      float v = st[cl * 129 + l];
      v = fmaxf(fmaf(v, sscale[cl], sshift[cl]), 0.f);
      h0[(((size_t)n * Ll + l) << 8) + cg * 32 + cl] = f2bf(v);
    }
  }
}

// ---------------- pack layer-1 weights into per-lane B fragments ------------
__global__ void pack_w1(const float* __restrict__ w0, const float* __restrict__ w1,
                        const float* __restrict__ w2, const float* __restrict__ w3,
                        unsigned short* __restrict__ wfrag) {
  int flat = blockIdx.x * 256 + threadIdx.x; // 262144 total
  int frag = flat >> 9;
  int within = flat & 511;
  int lane = within >> 4, j = within & 15;
  int base, k; const float* w;
  if (frag < 32)       { base = 0;   k = 1; w = w0; }
  else if (frag < 128) { base = 32;  k = 3; w = w1; }
  else if (frag < 288) { base = 128; k = 5; w = w2; }
  else                 { base = 288; k = 7; w = w3; }
  int rel = frag - base;
  int kt = rel & 7;
  int ctt = rel >> 3;
  int t = ctt % k, ct = ctt / k;
  int ncol = lane & 15, khalf = lane >> 4;
  int v = j >> 1, pos = j & 1;
  int kk = (v < 4) ? (v * 2 + pos + khalf * 8) : ((v - 4) * 2 + pos + 16 + khalf * 8);
  int c = kt * 32 + kk;
  int co = ct * 16 + ncol;
  wfrag[flat] = f2bf(w[((size_t)co * Dd + c) * k + t]);
}

// ---------------- Layer 1: implicit-im2col conv via bf16 WMMA ---------------
// block = (n, half-of-L). Wave w owns 2 co-tiles (one k in {1,3}, one k in
// {5,7} -> all waves do 256 WMMAs) across all 4 m-tiles, so every global
// B-fragment load feeds 4 WMMAs. h0 tile staged into LDS by the TDM.
__global__ void l1_conv(const unsigned short* __restrict__ h0,
                        const unsigned short* __restrict__ wfrag,
                        float* __restrict__ y) {
  __shared__ u32x4 hs4[2240]; // 70 rows (l-3..l+66) x 256 ch bf16
  const int n = blockIdx.x >> 1;
  const int lhalf = blockIdx.x & 1;
  const int tid = threadIdx.x;
  const int wid = tid >> 5, lane = tid & 31;
  const int col = lane & 15, khalf = lane >> 4;

  // zero the 3 halo rows that fall outside [0,128)
  unsigned int* hsu = (unsigned int*)hs4;
  const int zbase = (lhalf == 0) ? 0 : 67 * 128;
  for (int idx = tid; idx < 384; idx += 256) hsu[zbase + idx] = 0u;

  // Tensor Data Mover: DMA 67 rows x 256 bf16 into LDS (wave 0 issues one D#)
  if (wid == 0) {
    const int row0 = lhalf * 64 - 3;
    const int gl0 = (row0 < 0) ? 0 : row0;
    const int ldsrow0 = gl0 - row0; // 3 for lhalf==0 else 0
    unsigned long long gaddr =
        (unsigned long long)(const void*)(h0 + (((size_t)n * Ll + gl0) << 8));
    unsigned int laddr =
        (unsigned int)(uintptr_t)hsu + (unsigned int)(ldsrow0 * 512);
    unsigned int tdim1 = (unsigned int)(Nn * Ll) - (unsigned int)(n * Ll + gl0);
    u32x4 g0;
    g0[0] = 1u;                                   // count=1, user descriptor
    g0[1] = laddr;                                // lds_addr (bytes)
    g0[2] = (unsigned int)gaddr;                  // global_addr[31:0]
    g0[3] = (unsigned int)((gaddr >> 32) & 0x01ffffffu) | (2u << 30); // [56:32]|type=2
    i32x8 g1;
    g1[0] = (int)(1u << 16);                      // data_size = 2B
    g1[1] = (int)(256u << 16);                    // tensor_dim0 = 256
    g1[2] = (int)((tdim1 & 0xffffu) << 16);       // tensor_dim1 lo16
    g1[3] = (int)((tdim1 >> 16) & 0xffffu) | (int)(256u << 16); // tdim1 hi | tile_dim0=256
    g1[4] = 67;                                   // tile_dim1 = 67 rows, tile_dim2=0
    g1[5] = 256;                                  // tensor_dim0_stride = 256
    g1[6] = 0;
    g1[7] = 0;
    i32x4 gz4 = {0, 0, 0, 0};
    i32x8 gz8 = {0, 0, 0, 0, 0, 0, 0, 0};
    __builtin_amdgcn_tensor_load_to_lds(g0, g1, gz4, gz4, gz8, 0);
    __builtin_amdgcn_s_wait_tensorcnt(0);
  }
  __syncthreads();

  v8f acc[8] = {};
  const u32x4* wf4 = (const u32x4*)wfrag;
  const int ct = wid & 3;
  const int lo = (wid < 4) ? 1 : 0;
#pragma unroll
  for (int s = 0; s < 2; ++s) {
    int k  = (s == 0) ? (lo ? 1 : 3) : (lo ? 7 : 5);
    int fb = (s == 0) ? (lo ? 0 : 32) : (lo ? 288 : 128);
    int pad = (k - 1) >> 1;
    for (int t = 0; t < k; ++t) {
      for (int kt = 0; kt < 8; ++kt) {
        int frag = fb + ((ct * k + t) * 8 + kt);
        union { u32x4 u[2]; v16bf v; } bu;
        const u32x4* bp = wf4 + ((size_t)frag * 32 + lane * 2);
        bu.u[0] = bp[0]; bu.u[1] = bp[1];
#pragma unroll
        for (int m = 0; m < 4; ++m) {
          int lds_row = m * 16 + col + t - pad + 3; // always 0..69
          union { u32x4 u[2]; v16bf v; } au;
          int base4 = lds_row * 32 + kt * 4 + khalf;
          au.u[0] = hs4[base4];
          au.u[1] = hs4[base4 + 2];
          acc[m * 2 + s] = __builtin_amdgcn_wmma_f32_16x16x32_bf16(
              false, au.v, false, bu.v, (short)0, acc[m * 2 + s], false, false);
        }
      }
    }
  }
#pragma unroll
  for (int i = 0; i < 8; i++) {
    int m = i >> 1, s = i & 1;
    int ci = (s == 0) ? (lo ? 0 : 1) : (lo ? 3 : 2);
    int c = (ci * 4 + ct) * 16 + col;
    float* dst = y + (((size_t)n * Dd + c) << 7) + (lhalf * 64 + m * 16 + khalf * 8);
    f32x4 v0 = {acc[i][0], acc[i][1], acc[i][2], acc[i][3]};
    f32x4 v1 = {acc[i][4], acc[i][5], acc[i][6], acc[i][7]};
    *(f32x4*)dst = v0;
    *(f32x4*)(dst + 4) = v1;
  }
}

// ---------------- BN+ReLU fused with mean over L ----------------------------
__global__ void l1_apply_mean(const float* __restrict__ y, const float* __restrict__ stats,
                              const float* g0, const float* g1, const float* g2, const float* g3,
                              const float* b0, const float* b1, const float* b2, const float* b3,
                              float* __restrict__ hidden) {
  const int n = blockIdx.x, c = threadIdx.x;
  const float cnt = (float)(Nn * Ll);
  float mean = stats[c] / cnt;
  float var = stats[Dd + c] / cnt - mean * mean;
  int ci = c >> 6, cidx = c & 63;
  const float* gp = (ci == 0) ? g0 : (ci == 1) ? g1 : (ci == 2) ? g2 : g3;
  const float* bp = (ci == 0) ? b0 : (ci == 1) ? b1 : (ci == 2) ? b2 : b3;
  float sc = gp[cidx] * rsqrtf(var + EPSf);
  float sh = bp[cidx] - mean * sc;
  const f32x4* p = (const f32x4*)(y + (((size_t)n * Dd + c) << 7));
  float s = 0.f;
  for (int l4 = 0; l4 < 32; l4++) {
    f32x4 v = p[l4];
    s += fmaxf(fmaf(v.x, sc, sh), 0.f) + fmaxf(fmaf(v.y, sc, sh), 0.f) +
         fmaxf(fmaf(v.z, sc, sh), 0.f) + fmaxf(fmaf(v.w, sc, sh), 0.f);
  }
  hidden[(size_t)n * Dd + c] = s * (1.f / 128.f);
}

// ---------------- small dense: out[i,j] = b[j] + in[i,:]·w[j,:] -------------
__global__ void dense(const float* __restrict__ in, const float* __restrict__ w,
                      const float* __restrict__ b, float* __restrict__ out,
                      int Jdim, int K, int act) {
  const int i = blockIdx.x;
  for (int j = threadIdx.x; j < Jdim; j += blockDim.x) {
    float acc = b[j];
    const float* ip = in + (size_t)i * K;
    const float* wp = w + (size_t)j * K;
    for (int kk = 0; kk < K; kk += 4) {
      acc = fmaf(ip[kk], wp[kk], acc);
      acc = fmaf(ip[kk + 1], wp[kk + 1], acc);
      acc = fmaf(ip[kk + 2], wp[kk + 2], acc);
      acc = fmaf(ip[kk + 3], wp[kk + 3], acc);
    }
    if (act) acc = leaky(acc);
    out[(size_t)i * Jdim + j] = acc;
  }
}

__global__ void s_kernel(const float* __restrict__ x1, const float* __restrict__ a,
                         float* __restrict__ s1, float* __restrict__ s2) {
  const int i = blockIdx.x * 256 + threadIdx.x;
  const float* xp = x1 + (size_t)i * Dd;
  float v1 = 0.f, v2 = 0.f;
  for (int kk = 0; kk < Dd; kk++) {
    float xv = xp[kk];
    v1 = fmaf(xv, a[kk], v1);
    v2 = fmaf(xv, a[Dd + kk], v2);
  }
  s1[i] = v1; s2[i] = v2;
}

__global__ void s1max_kernel(const float* __restrict__ s1, float* __restrict__ outmax) {
  __shared__ float sm[256];
  const int tid = threadIdx.x;
  float m = -1e30f;
  for (int i = tid; i < Nn; i += 256) m = fmaxf(m, s1[i]);
  sm[tid] = m; __syncthreads();
  for (int off = 128; off > 0; off >>= 1) {
    if (tid < off) sm[tid] = fmaxf(sm[tid], sm[tid + off]);
    __syncthreads();
  }
  if (tid == 0) outmax[0] = sm[0];
}

// ---------------- streaming softmax attention @ hidden, f32 WMMA ------------
__global__ void attn_kernel(const float* __restrict__ s1, const float* __restrict__ s2,
                            const float* __restrict__ s1maxp, const float* __restrict__ hidden,
                            float* __restrict__ hidden2) {
  __shared__ float ps[16 * 132];
  __shared__ float sdenom[16];
  __shared__ float smax[16];
  __shared__ float ss2[16];
  const int i0 = blockIdx.x * 16;
  const int tid = threadIdx.x;
  if (tid < 16) {
    float v = s2[i0 + tid];
    ss2[tid] = v;
    smax[tid] = leaky(v + s1maxp[0]);
    sdenom[tid] = 0.f;
  }
  __syncthreads();
  const int lane = tid & 31, wid = tid >> 5;
  const int ct0 = wid * 2;
  const int col = lane & 15, khalf = lane >> 4;
  v8f acc0 = {}, acc1 = {};
  for (int j0 = 0; j0 < Nn; j0 += 128) {
    for (int it = 0; it < 8; ++it) {
      int idx = tid + it * 256;
      int r = idx >> 7, jj = idx & 127;
      float p = __expf(leaky(ss2[r] + s1[j0 + jj]) - smax[r]);
      ps[r * 132 + jj] = p;
      atomicAdd(&sdenom[r], p);
    }
    __syncthreads();
    for (int kt = 0; kt < 32; ++kt) {
      f32x2 a = *(const f32x2*)&ps[(lane & 15) * 132 + kt * 4 + khalf * 2];
      int jrow = j0 + kt * 4 + khalf * 2;
      {
        int c = ct0 * 16 + col;
        f32x2 b;
        b.x = hidden[(size_t)jrow * Dd + c];
        b.y = hidden[(size_t)(jrow + 1) * Dd + c];
        acc0 = __builtin_amdgcn_wmma_f32_16x16x4_f32(false, a, false, b, (short)0, acc0, false, false);
      }
      {
        int c = (ct0 + 1) * 16 + col;
        f32x2 b;
        b.x = hidden[(size_t)jrow * Dd + c];
        b.y = hidden[(size_t)(jrow + 1) * Dd + c];
        acc1 = __builtin_amdgcn_wmma_f32_16x16x4_f32(false, a, false, b, (short)0, acc1, false, false);
      }
    }
    __syncthreads();
  }
#pragma unroll
  for (int j = 0; j < 8; j++) {
    int m = khalf * 8 + j;
    float inv = 1.f / sdenom[m];
    {
      int c = ct0 * 16 + col;
      size_t o = (size_t)(i0 + m) * Dd + c;
      hidden2[o] = acc0[j] * inv + hidden[o];
    }
    {
      int c = (ct0 + 1) * 16 + col;
      size_t o = (size_t)(i0 + m) * Dd + c;
      hidden2[o] = acc1[j] * inv + hidden[o];
    }
  }
}

extern "C" void kernel_launch(void* const* d_in, const int* in_sizes, int n_in,
                              void* d_out, int out_size, void* d_ws, size_t ws_size,
                              hipStream_t stream) {
  const float* x = (const float*)d_in[0];
  const float *l0w[4], *l0g[4], *l0bt[4], *l1w[4], *l1g[4], *l1bt[4];
  for (int ci = 0; ci < 4; ci++) {
    l0w[ci] = (const float*)d_in[1 + 4 * ci];
    l0g[ci] = (const float*)d_in[3 + 4 * ci];
    l0bt[ci] = (const float*)d_in[4 + 4 * ci];
    l1w[ci] = (const float*)d_in[17 + 4 * ci];
    l1g[ci] = (const float*)d_in[19 + 4 * ci];
    l1bt[ci] = (const float*)d_in[20 + 4 * ci];
  }
  const float* trans_w = (const float*)d_in[33];
  const float* trans_b = (const float*)d_in[34];
  const float* a = (const float*)d_in[35];
  const float* fc_w = (const float*)d_in[36];
  const float* fc_b = (const float*)d_in[37];
  const float* g1_w = (const float*)d_in[38];
  const float* g1_b = (const float*)d_in[39];
  const float* g2_w = (const float*)d_in[40];
  const float* g2_b = (const float*)d_in[41];

  char* ws = (char*)d_ws;
  float* y = (float*)ws;                                       // 512 MB raw conv out (reused)
  unsigned short* h0 = (unsigned short*)(ws + 536870912ull);   // 256 MB bf16 [n][l][c]
  unsigned short* wfrag = (unsigned short*)(ws + 805306368ull);// 512 KB packed B frags
  float* stats0 = (float*)(ws + 805830656ull);
  float* stats1 = (float*)(ws + 805832704ull);
  float* hidden = (float*)(ws + 805834752ull);                 // 4 MB
  float* x1 = (float*)(ws + 810029056ull);                     // 4 MB (reused as z)
  float* s1 = (float*)(ws + 814223360ull);
  float* s2 = (float*)(ws + 814239744ull);
  float* s1m = (float*)(ws + 814256128ull);
  float* z = x1;

  float* origin_out = (float*)d_out;
  float* gat_out = origin_out + (size_t)Nn * NCLASSc;
  float* hidden2 = gat_out + (size_t)Nn * NCLASSc;

  l0_conv<<<Nn, 128, 0, stream>>>(x, l0w[0], l0w[1], l0w[2], l0w[3], y);
  pack_w1<<<1024, 256, 0, stream>>>(l1w[0], l1w[1], l1w[2], l1w[3], wfrag);
  bn_stats<<<Dd, 256, 0, stream>>>(y, stats0);
  l0_apply<<<Nn, 256, 0, stream>>>(y, stats0, l0g[0], l0g[1], l0g[2], l0g[3],
                                   l0bt[0], l0bt[1], l0bt[2], l0bt[3], h0);
  l1_conv<<<Nn * 2, 256, 0, stream>>>(h0, wfrag, y);
  bn_stats<<<Dd, 256, 0, stream>>>(y, stats1);
  l1_apply_mean<<<Nn, 256, 0, stream>>>(y, stats1, l1g[0], l1g[1], l1g[2], l1g[3],
                                        l1bt[0], l1bt[1], l1bt[2], l1bt[3], hidden);
  dense<<<Nn, 64, 0, stream>>>(hidden, fc_w, fc_b, origin_out, NCLASSc, Dd, 0);
  dense<<<Nn, 256, 0, stream>>>(hidden, trans_w, trans_b, x1, Dd, Dd, 0);
  s_kernel<<<Nn / 256, 256, 0, stream>>>(x1, a, s1, s2);
  s1max_kernel<<<1, 256, 0, stream>>>(s1, s1m);
  attn_kernel<<<Nn / 16, 256, 0, stream>>>(s1, s2, s1m, hidden, hidden2);
  dense<<<Nn, 256, 0, stream>>>(hidden2, g1_w, g1_b, z, Dd, Dd, 1);
  dense<<<Nn, 64, 0, stream>>>(z, g2_w, g2_b, gat_out, NCLASSc, Dd, 0);
}
